// LSTMProjStateModel_88012469830030
// MI455X (gfx1250) — matrix operand order repaired and checked by reading, not compiled, not run on hardware
//
#include <hip/hip_runtime.h>
#include <math.h>

// ---------------- types for WMMA fragments ----------------
typedef __attribute__((ext_vector_type(16))) __bf16 v16bf;
typedef __attribute__((ext_vector_type(8)))  __bf16 v8bf;
typedef __attribute__((ext_vector_type(8)))  float  v8f;

typedef int v4i_ __attribute__((vector_size(16)));

union FragU { v16bf v; v8bf h[2]; };

#define LDT 40   // padded LDS stride (elements) for 32-wide K tiles: 80B = 5*16B

static constexpr int T_ = 256, B_ = 128, I_ = 1024, H_ = 1024, P_ = 512, G_ = 4096;

#define WMMA_BF16(a, b, c) \
  __builtin_amdgcn_wmma_f32_16x16x32_bf16(false, (a), false, (b), (short)0, (c), false, false)

// ---------------- CDNA5 async global->LDS copy (ASYNCcnt-tracked DMA) ----------------
static __device__ __forceinline__ void async_cp16(void* lds_dst, const void* gsrc) {
#if __has_builtin(__builtin_amdgcn_global_load_async_to_lds_b128)
  __builtin_amdgcn_global_load_async_to_lds_b128(
      (__attribute__((address_space(1))) v4i_*)gsrc,
      (__attribute__((address_space(3))) v4i_*)lds_dst, 0, 0);
#else
  // LDS generic addresses carry the byte offset in the low 32 bits (aperture rule)
  unsigned int       l = (unsigned int)(unsigned long long)(uintptr_t)lds_dst;
  unsigned long long g = (unsigned long long)(uintptr_t)gsrc;
  asm volatile("global_load_async_to_lds_b128 %0, %1, off" :: "v"(l), "v"(g) : "memory");
#endif
}

static __device__ __forceinline__ void async_wait0() {
#if __has_builtin(__builtin_amdgcn_s_wait_asynccnt)
  __builtin_amdgcn_s_wait_asynccnt(0);
#else
  asm volatile("s_wait_asynccnt 0x0" ::: "memory");
#endif
}

// A-fragment: 16x32 bf16, lane = (hf, m). elems 0..7 = K hf*8.., elems 8..15 = K 16+hf*8..
static __device__ __forceinline__ v16bf frag_a(const __bf16* row, int hf) {
  FragU u;
  u.h[0] = *(const v8bf*)(row + hf * 8);
  u.h[1] = *(const v8bf*)(row + 16 + hf * 8);
  return u.v;
}
// B-fragment: 32x16 bf16 stored [n][k] in LDS: elems 0..7 = K hf*8.., elems 8..15 = K 16+hf*8..
static __device__ __forceinline__ v16bf frag_b(const __bf16* row, int hf) {
  FragU u;
  u.h[0] = *(const v8bf*)(row + hf * 8);
  u.h[1] = *(const v8bf*)(row + 16 + hf * 8);
  return u.v;
}

static __device__ __forceinline__ float sigmoidf_(float x) {
  return 1.0f / (1.0f + expf(-x));
}

// ---------------- utility kernels ----------------
__global__ void k_f32_to_bf16(const float* __restrict__ in, __bf16* __restrict__ out, long n) {
  long i = (long)blockIdx.x * blockDim.x + threadIdx.x;
  long s = (long)gridDim.x * blockDim.x;
  for (; i < n; i += s) out[i] = (__bf16)in[i];
}

__global__ void k_f32_copy(const float* __restrict__ in, float* __restrict__ out, long n) {
  long i = (long)blockIdx.x * blockDim.x + threadIdx.x;
  long s = (long)gridDim.x * blockDim.x;
  for (; i < n; i += s) out[i] = in[i];
}

// writes hT (= ys[T-1]) and cT into the tail of d_out
__global__ void k_tail(const float* __restrict__ ylast, const float* __restrict__ cbuf,
                       float* __restrict__ dst, int np, int nh) {
  int i = blockIdx.x * blockDim.x + threadIdx.x;
  if (i < np) dst[i] = ylast[i];
  if (i < nh) dst[np + i] = cbuf[i];
}

// ---------------- xg = x @ W_ih^T  :  [32768 x 1024] x [1024 x 4096] ----------------
// block tile 128x128, 8 waves, wave tile 32x64 (2x4 accum tiles), K step 32
// double-buffered LDS, async global->LDS staging overlapped with WMMA
__global__ __launch_bounds__(256) void xg_gemm(const __bf16* __restrict__ X,   // [M][K=1024]
                                               const __bf16* __restrict__ W,   // [N=4096][K]
                                               float* __restrict__ XG) {       // [M][N]
  const int K = I_, N = G_;
  __shared__ __align__(16) __bf16 As[2][128 * LDT];
  __shared__ __align__(16) __bf16 Bs[2][128 * LDT];

  const int m0 = blockIdx.x * 128;
  const int n0 = blockIdx.y * 128;
  const int tid = threadIdx.x;
  const int lane = tid & 31, w = tid >> 5;
  const int wm = w >> 1, wn = w & 1;       // 4x2 wave grid
  const int hf = lane >> 4, lm = lane & 15;

  v8f acc[2][4];
  const v8f vz = {0.f, 0.f, 0.f, 0.f, 0.f, 0.f, 0.f, 0.f};
#pragma unroll
  for (int i = 0; i < 2; ++i)
#pragma unroll
    for (int j = 0; j < 4; ++j) acc[i][j] = vz;

  const int sr = tid >> 2, sc = tid & 3;   // staging row/chunk

  // prologue: async-stage first K tile into buffer 0
#pragma unroll
  for (int rr = 0; rr < 128; rr += 64) {
    async_cp16(&As[0][(rr + sr) * LDT + sc * 8], X + (size_t)(m0 + rr + sr) * K + sc * 8);
    async_cp16(&Bs[0][(rr + sr) * LDT + sc * 8], W + (size_t)(n0 + rr + sr) * K + sc * 8);
  }
  async_wait0();
  __syncthreads();

  int buf = 0;
  for (int k0 = 0; k0 < K; k0 += 32) {
    const int nbuf = buf ^ 1;
    if (k0 + 32 < K) {  // async-stage next K tile while computing this one
#pragma unroll
      for (int rr = 0; rr < 128; rr += 64) {
        async_cp16(&As[nbuf][(rr + sr) * LDT + sc * 8],
                   X + (size_t)(m0 + rr + sr) * K + (k0 + 32) + sc * 8);
        async_cp16(&Bs[nbuf][(rr + sr) * LDT + sc * 8],
                   W + (size_t)(n0 + rr + sr) * K + (k0 + 32) + sc * 8);
      }
    }

    v16bf a[2], b[4];
#pragma unroll
    for (int mi = 0; mi < 2; ++mi) a[mi] = frag_a(&As[buf][(wm * 32 + mi * 16 + lm) * LDT], hf);
#pragma unroll
    for (int ni = 0; ni < 4; ++ni) b[ni] = frag_b(&Bs[buf][(wn * 64 + ni * 16 + lm) * LDT], hf);
#pragma unroll
    for (int mi = 0; mi < 2; ++mi)
#pragma unroll
      for (int ni = 0; ni < 4; ++ni) acc[mi][ni] = WMMA_BF16(a[mi], b[ni], acc[mi][ni]);

    async_wait0();     // next tile landed in LDS
    __syncthreads();   // all waves done reading `buf` and all async writes visible
    buf = nbuf;
  }

#pragma unroll
  for (int mi = 0; mi < 2; ++mi)
#pragma unroll
    for (int ni = 0; ni < 4; ++ni)
#pragma unroll
      for (int r = 0; r < 8; ++r) {
        int row = m0 + wm * 32 + mi * 16 + hf * 8 + r;
        int col = n0 + wn * 64 + ni * 16 + lm;
        XG[(size_t)row * N + col] = acc[mi][ni][r];
      }
}

// ---------------- per-step gates + activations ----------------
// gates[b, gate*H + j] = xg_t + h @ W_hh^T ; then LSTM pointwise -> c_new, h_full(bf16)
// grid = H/64 blocks of 512 threads (16 waves); block covers all B rows, 4 gates x 64 cols
__global__ __launch_bounds__(512) void gates_step(const __bf16* __restrict__ Hb,   // [128][512]
                                                  const __bf16* __restrict__ Whh,  // [4096][512]
                                                  const float* __restrict__ XGt,   // [128][4096]
                                                  float* __restrict__ Cbuf,        // [128][1024]
                                                  __bf16* __restrict__ Hfull) {    // [128][1024]
  const int K = P_;
  const int j0 = blockIdx.x * 64;
  __shared__ __align__(16) __bf16 As[128 * LDT];
  __shared__ __align__(16) __bf16 Bs[256 * LDT];
  __shared__ float Gs[128 * 256];  // [b][gate*64 + j]

  const int tid = threadIdx.x;
  const int lane = tid & 31, w = tid >> 5;       // 16 waves
  const int wm = w & 3;                          // row group (4 x 32)
  const int wg = w >> 2;                         // gate 0..3
  const int hf = lane >> 4, lm = lane & 15;

  v8f acc[2][4];
  const v8f vz = {0.f, 0.f, 0.f, 0.f, 0.f, 0.f, 0.f, 0.f};
#pragma unroll
  for (int i = 0; i < 2; ++i)
#pragma unroll
    for (int j = 0; j < 4; ++j) acc[i][j] = vz;

  for (int k0 = 0; k0 < K; k0 += 32) {
    {  // A: 128 rows x 4 chunks == 512 threads
      int r = tid >> 2, c = tid & 3;
      async_cp16(As + r * LDT + c * 8, Hb + (size_t)r * K + k0 + c * 8);
    }
#pragma unroll
    for (int it = 0; it < 2; ++it) {  // B: 256 rows x 4 chunks, 2 per thread
      int idx = tid + it * 512;
      int r = idx >> 2, c = idx & 3;
      int gate = r >> 6, jr = r & 63;
      async_cp16(Bs + r * LDT + c * 8,
                 Whh + (size_t)(gate * H_ + j0 + jr) * K + k0 + c * 8);
    }
    async_wait0();
    __syncthreads();

    v16bf a[2], b[4];
#pragma unroll
    for (int mi = 0; mi < 2; ++mi) a[mi] = frag_a(As + (wm * 32 + mi * 16 + lm) * LDT, hf);
#pragma unroll
    for (int ni = 0; ni < 4; ++ni) b[ni] = frag_b(Bs + (wg * 64 + ni * 16 + lm) * LDT, hf);
#pragma unroll
    for (int mi = 0; mi < 2; ++mi)
#pragma unroll
      for (int ni = 0; ni < 4; ++ni) acc[mi][ni] = WMMA_BF16(a[mi], b[ni], acc[mi][ni]);
    __syncthreads();
  }

  // epilogue: add xg, deposit gate tiles to LDS for cross-wave exchange
#pragma unroll
  for (int mi = 0; mi < 2; ++mi)
#pragma unroll
    for (int ni = 0; ni < 4; ++ni)
#pragma unroll
      for (int r = 0; r < 8; ++r) {
        int row = wm * 32 + mi * 16 + hf * 8 + r;  // batch index
        int colg = ni * 16 + lm;                   // 0..63 within gate
        float v = acc[mi][ni][r] + XGt[(size_t)row * G_ + wg * H_ + j0 + colg];
        Gs[row * 256 + wg * 64 + colg] = v;
      }
  __syncthreads();

  // LSTM pointwise over this block's (b, j) footprint
  for (int idx = tid; idx < 128 * 64; idx += 512) {
    int b = idx >> 6, j = idx & 63;
    float gi = Gs[b * 256 + j];
    float gf = Gs[b * 256 + 64 + j];
    float gg = Gs[b * 256 + 128 + j];
    float go = Gs[b * 256 + 192 + j];
    float iv = sigmoidf_(gi);
    float fv = sigmoidf_(gf);
    float gv = tanhf(gg);
    float ov = sigmoidf_(go);
    int col = j0 + j;
    float cn = fv * Cbuf[b * H_ + col] + iv * gv;
    Cbuf[b * H_ + col] = cn;
    Hfull[b * H_ + col] = (__bf16)(ov * tanhf(cn));
  }
}

// ---------------- per-step projection: h_new = h_full @ W_hr^T ----------------
// [128 x 1024] x [1024 x 512]; writes ys[t] (f32) and next h (bf16)
// grid = 512/64 blocks of 256 threads (8 waves), wave tile 32x32 (2x2 accum)
__global__ __launch_bounds__(256) void proj_step(const __bf16* __restrict__ Hfull,  // [128][1024]
                                                 const __bf16* __restrict__ Whr,    // [512][1024]
                                                 float* __restrict__ Yt,            // [128][512]
                                                 __bf16* __restrict__ Hb) {         // [128][512]
  const int K = H_, N = P_;
  const int n0 = blockIdx.x * 64;
  __shared__ __align__(16) __bf16 As[128 * LDT];
  __shared__ __align__(16) __bf16 Bs[64 * LDT];

  const int tid = threadIdx.x;
  const int lane = tid & 31, w = tid >> 5;  // 8 waves
  const int wm = w & 3;                     // rows (4 x 32)
  const int wn = w >> 2;                    // cols (2 x 32)
  const int hf = lane >> 4, lm = lane & 15;

  v8f acc[2][2];
  const v8f vz = {0.f, 0.f, 0.f, 0.f, 0.f, 0.f, 0.f, 0.f};
#pragma unroll
  for (int i = 0; i < 2; ++i)
#pragma unroll
    for (int j = 0; j < 2; ++j) acc[i][j] = vz;

  for (int k0 = 0; k0 < K; k0 += 32) {
#pragma unroll
    for (int it = 0; it < 2; ++it) {  // A: 128 rows x 4 chunks, 2 per thread
      int idx = tid + it * 256;
      int r = idx >> 2, c = idx & 3;
      async_cp16(As + r * LDT + c * 8, Hfull + (size_t)r * K + k0 + c * 8);
    }
    {  // B: 64 rows x 4 chunks == 256 threads
      int r = tid >> 2, c = tid & 3;
      async_cp16(Bs + r * LDT + c * 8, Whr + (size_t)(n0 + r) * K + k0 + c * 8);
    }
    async_wait0();
    __syncthreads();

    v16bf a[2], b[2];
#pragma unroll
    for (int mi = 0; mi < 2; ++mi) a[mi] = frag_a(As + (wm * 32 + mi * 16 + lm) * LDT, hf);
#pragma unroll
    for (int ni = 0; ni < 2; ++ni) b[ni] = frag_b(Bs + (wn * 32 + ni * 16 + lm) * LDT, hf);
#pragma unroll
    for (int mi = 0; mi < 2; ++mi)
#pragma unroll
      for (int ni = 0; ni < 2; ++ni) acc[mi][ni] = WMMA_BF16(a[mi], b[ni], acc[mi][ni]);
    __syncthreads();
  }

#pragma unroll
  for (int mi = 0; mi < 2; ++mi)
#pragma unroll
    for (int ni = 0; ni < 2; ++ni)
#pragma unroll
      for (int r = 0; r < 8; ++r) {
        int row = wm * 32 + mi * 16 + hf * 8 + r;
        int col = n0 + wn * 32 + ni * 16 + lm;
        float v = acc[mi][ni][r];
        Yt[(size_t)row * N + col] = v;
        Hb[(size_t)row * N + col] = (__bf16)v;
      }
}

// ---------------- host orchestration ----------------
static inline size_t alignup(size_t x) { return (x + 255) & ~(size_t)255; }

extern "C" void kernel_launch(void* const* d_in, const int* in_sizes, int n_in,
                              void* d_out, int out_size, void* d_ws, size_t ws_size,
                              hipStream_t stream) {
  const float* x   = (const float*)d_in[0];  // [T,B,I]
  const float* h0  = (const float*)d_in[1];  // [1,B,P]
  const float* c0  = (const float*)d_in[2];  // [1,B,H]
  const float* Wih = (const float*)d_in[3];  // [4H,I]
  const float* Whh = (const float*)d_in[4];  // [4H,P]
  const float* Whr = (const float*)d_in[5];  // [P,H]
  float* out = (float*)d_out;

  const size_t nx = (size_t)T_ * B_ * I_;
  const size_t nwih = (size_t)4 * H_ * I_;
  const size_t nwhh = (size_t)4 * H_ * P_;
  const size_t nwhr = (size_t)P_ * H_;

  char* p = (char*)d_ws;
  __bf16* xbf  = (__bf16*)p; p += alignup(nx * 2);
  __bf16* wihb = (__bf16*)p; p += alignup(nwih * 2);
  __bf16* whhb = (__bf16*)p; p += alignup(nwhh * 2);
  __bf16* whrb = (__bf16*)p; p += alignup(nwhr * 2);
  __bf16* hbuf = (__bf16*)p; p += alignup((size_t)B_ * P_ * 2);
  __bf16* hfull = (__bf16*)p; p += alignup((size_t)B_ * H_ * 2);
  float*  cbuf = (float*)p;  p += alignup((size_t)B_ * H_ * 4);
  float*  xg   = (float*)p;  p += alignup((size_t)T_ * B_ * G_ * 4);

  // one-time (per call) conversions to bf16 + state init — keeps call deterministic
  k_f32_to_bf16<<<4096, 256, 0, stream>>>(x, xbf, (long)nx);
  k_f32_to_bf16<<<2048, 256, 0, stream>>>(Wih, wihb, (long)nwih);
  k_f32_to_bf16<<<1024, 256, 0, stream>>>(Whh, whhb, (long)nwhh);
  k_f32_to_bf16<<<512, 256, 0, stream>>>(Whr, whrb, (long)nwhr);
  k_f32_to_bf16<<<256, 256, 0, stream>>>(h0, hbuf, (long)B_ * P_);
  k_f32_copy<<<512, 256, 0, stream>>>(c0, cbuf, (long)B_ * H_);

  // precompute xg = x @ W_ih^T   (bulk of the FLOPs, fully parallel)
  xg_gemm<<<dim3((T_ * B_) / 128, G_ / 128), 256, 0, stream>>>(xbf, wihb, xg);

  // sequential recurrence
  for (int t = 0; t < T_; ++t) {
    gates_step<<<H_ / 64, 512, 0, stream>>>(hbuf, whhb, xg + (size_t)t * B_ * G_, cbuf, hfull);
    proj_step<<<P_ / 64, 256, 0, stream>>>(hfull, whrb, out + (size_t)t * B_ * P_, hbuf);
  }

  // tail: hT = ys[T-1], cT = cbuf
  k_tail<<<512, 256, 0, stream>>>(out + (size_t)(T_ - 1) * B_ * P_, cbuf,
                                  out + (size_t)T_ * B_ * P_, B_ * P_, B_ * H_);
}
